// DFMBitFlipPredictor_88940182765939
// MI455X (gfx1250) — matrix operand order repaired and checked by reading, not compiled, not run on hardware
//
#include <hip/hip_runtime.h>
#include <hip/hip_bf16.h>
#include <math.h>

typedef __attribute__((ext_vector_type(16))) _Float16 v16h;
typedef __attribute__((ext_vector_type(8)))  float    v8f;
typedef __attribute__((ext_vector_type(4)))  _Float16 h4;
typedef __attribute__((ext_vector_type(2)))  _Float16 h2;
typedef __attribute__((ext_vector_type(4)))  unsigned int v4u;
typedef __attribute__((ext_vector_type(8)))  int      v8i;
typedef __attribute__((ext_vector_type(4)))  int      v4i;

#define Bn 4
#define Nnode 256
#define Hd 128
#define Ld 4
#define PITCH 136   // f16 row pitch for 16x128 LDS panels (+8 f16 pad vs banks)

__device__ __forceinline__ float silu_f(float x) {
  return x / (1.0f + __expf(-x));
}

// ------------------------------------------------------------------
// WMMA GEMM specialized to M=1024, K=128, Nc=128 (every projection here).
// C = act(A1)@B1 (+ A2@B2) + bias_scale*bias + resid
// One wave per 16x16 tile. Data path: coalesced global_load_b128 ->
// f16 LDS panels (B stored transposed) -> 32-bit DS reads in the exact
// CDNA5 ISA 7.12.2 operand layouts -> v_wmma_f32_16x16x32_f16.
// ------------------------------------------------------------------
__global__ void gemm_wmma_f16(const float* __restrict__ A1,
                              const float* __restrict__ B1,
                              const float* __restrict__ A2,
                              const float* __restrict__ B2,
                              const float* __restrict__ bias, float bias_scale,
                              const float* __restrict__ resid,
                              float* __restrict__ C,
                              int preact_silu)
{
  __shared__ _Float16 lA[16 * PITCH];   // A panel: [m=16][k=128]
  __shared__ _Float16 lB[16 * PITCH];   // B^T panel: [n=16][k=128]

  const int lane = threadIdx.x;         // 0..31 (wave32)
  const int half = lane >> 4;
  const int l15  = lane & 15;
  const int tm   = blockIdx.x >> 3;     // Nc/16 = 8 tiles along N
  const int tn   = blockIdx.x & 7;
  const int row0 = tm << 4;
  const int col0 = tn << 4;

  __builtin_prefetch(B1 + (size_t)col0, 0, 3);          // global_prefetch_b8
  __builtin_prefetch(A1 + (size_t)row0 * Hd, 0, 3);

  const int nMat = A2 ? 2 : 1;
  v8f c = {};

  for (int mats = 0; mats < nMat; ++mats) {
    const float* A  = mats ? A2 : A1;
    const float* Bm = mats ? B2 : B1;

    // ---- stage A panel: pass p loads row p (128 f32) coalesced as b128 ----
#pragma unroll 4
    for (int p = 0; p < 16; ++p) {
      float4 v = *(const float4*)(A + (size_t)(row0 + p) * Hd + lane * 4);
      if (preact_silu && mats == 0) {
        v.x = silu_f(v.x); v.y = silu_f(v.y); v.z = silu_f(v.z); v.w = silu_f(v.w);
      }
      h4 hv; hv[0] = (_Float16)v.x; hv[1] = (_Float16)v.y;
             hv[2] = (_Float16)v.z; hv[3] = (_Float16)v.w;
      *(h4*)&lA[p * PITCH + lane * 4] = hv;
    }
    // ---- stage B panel transposed: k rows 0..127, cols col0..col0+15 ----
    {
      const int q = lane >> 2, rem = lane & 3;
#pragma unroll 4
      for (int p = 0; p < 16; ++p) {
        int k  = p * 8 + q;
        int cc = rem * 4;
        float4 v = *(const float4*)(Bm + (size_t)k * Hd + col0 + cc);
        lB[(cc + 0) * PITCH + k] = (_Float16)v.x;
        lB[(cc + 1) * PITCH + k] = (_Float16)v.y;
        lB[(cc + 2) * PITCH + k] = (_Float16)v.z;
        lB[(cc + 3) * PITCH + k] = (_Float16)v.w;
      }
    }
    __syncthreads();   // single wave: DS in-order, but keep for clarity

    // ---- 4 WMMA steps over K=128 ----
#pragma unroll
    for (int kt = 0; kt < Hd; kt += 32) {
      v16h a, b;
#pragma unroll
      for (int ep = 0; ep < 8; ++ep) {
        int e  = ep << 1;
        // A 16x32 layout: k = e + 8*half (+8 if e>=8); pairs are k,k+1
        int ka = kt + e + (half << 3) + ((e >= 8) ? 8 : 0);
        h2 pa  = *(const h2*)&lA[l15 * PITCH + ka];
        a[e] = pa[0]; a[e + 1] = pa[1];
        // B 32x16 layout: k = e + 8*half (+8 if e>=8), n = lane&15; pairs are k,k+1
        int kb = kt + e + (half << 3) + ((e >= 8) ? 8 : 0);
        h2 pb  = *(const h2*)&lB[l15 * PITCH + kb];
        b[e] = pb[0]; b[e + 1] = pb[1];
      }
      c = __builtin_amdgcn_wmma_f32_16x16x32_f16(false, a, false, b,
                                                 (short)0, c, false, false);
    }
    __syncthreads();
  }

  // C/D layout: VGPR r -> row = r + 8*half, col = lane&15
#pragma unroll
  for (int r = 0; r < 8; ++r) {
    int row = row0 + r + (half << 3);
    int col = col0 + l15;
    float v = c[r];
    if (bias)  v += bias_scale * bias[col];
    if (resid) v += resid[(size_t)row * Hd + col];
    C[(size_t)row * Hd + col] = v;
  }
}

// h[b,n,:] = x_t[b,n]*W[0,:] + h_field[n]*W[1,:] + b[:]
__global__ void init_h_kernel(const float* __restrict__ x_t,
                              const float* __restrict__ h_field,
                              const float* __restrict__ npw,
                              const float* __restrict__ npb,
                              float* __restrict__ h)
{
  int idx = blockIdx.x * blockDim.x + threadIdx.x;   // B*N*H
  int hh = idx & (Hd - 1);
  int n  = (idx >> 7) & (Nnode - 1);
  int bn = idx >> 7;
  h[idx] = x_t[bn] * npw[hh] + h_field[n] * npw[Hd + hh] + npb[hh];
}

// gemb[b,:] = silu([t,beta] @ ge_w1 + b1) @ ge_w2 + b2
__global__ void gemb_kernel(const float* __restrict__ t,
                            const float* __restrict__ beta,
                            const float* __restrict__ w1,
                            const float* __restrict__ b1,
                            const float* __restrict__ w2,
                            const float* __restrict__ b2,
                            float* __restrict__ gemb)
{
  __shared__ float red[Hd];
  int hh = threadIdx.x;
  for (int b = 0; b < Bn; ++b) {
    float hid = silu_f(t[b] * w1[hh] + beta[b] * w1[Hd + hh] + b1[hh]);
    for (int d = 0; d < 2; ++d) {
      red[hh] = hid * w2[hh * 2 + d];
      __syncthreads();
      for (int s = Hd / 2; s > 0; s >>= 1) {
        if (hh < s) red[hh] += red[hh + s];
        __syncthreads();
      }
      if (hh == 0) gemb[b * 2 + d] = red[0] + b2[d];
      __syncthreads();
    }
  }
}

// S[b,i,:] = sum_j silu(ai[b,i,:] + aj[b,j,:] + J[i,j]*We[:])
// J row staged into LDS by the Tensor Data Mover (256x1 row tile, 4B elems).
__global__ void msg_sum_kernel(const float* __restrict__ ai,
                               const float* __restrict__ aj,
                               const float* __restrict__ J,
                               const float* __restrict__ We,
                               float* __restrict__ S)
{
  __shared__ float jrow[Nnode];        // 1 KB, filled by TDM
  int bi = blockIdx.x;                 // b*N + i
  int i  = bi & (Nnode - 1);
  int b  = bi >> 8;
  int hh = threadIdx.x;                // 0..127

  if (hh < 32) {                       // wave 0 issues the TDM descriptor
    unsigned long long gaddr = (unsigned long long)(const void*)(J + (size_t)i * Nnode);
    unsigned int       laddr = (unsigned int)(unsigned long long)(const void*)&jrow[0];
    // D# group0: count=1 | lds_addr | global_addr[56:0] | type=2
    v4u g0;
    g0[0] = 1u;
    g0[1] = laddr;
    g0[2] = (unsigned int)gaddr;
    g0[3] = (unsigned int)(gaddr >> 32) | (2u << 30);
    // D# group1: wg_mask=0, data_size=4B, tensor_dim0=256, tensor_dim1=1,
    //            tile_dim0=256, tile_dim1=1, tensor_dim0_stride=256
    v8i g1;
    g1[0] = 0x00020000;                 // data_size=2 (4 bytes)
    g1[1] = (int)(256u << 16);          // tensor_dim0[15:0] @ bits 63:48
    g1[2] = (int)(1u << 16);            // tensor_dim1[15:0] @ bits 95:80
    g1[3] = (int)(256u << 16);          // tile_dim0 @ bits 127:112
    g1[4] = 1;                          // tile_dim1 @ bits 143:128
    g1[5] = 256;                        // tensor_dim0_stride[31:0]
    g1[6] = 0;
    g1[7] = 0;
    v4i gz = {0, 0, 0, 0};
#if __clang_major__ >= 23
    v8i gz8 = {0, 0, 0, 0, 0, 0, 0, 0};
    __builtin_amdgcn_tensor_load_to_lds(g0, g1, gz, gz, gz8, 0);
#else
    __builtin_amdgcn_tensor_load_to_lds(g0, g1, gz, gz, 0);
#endif
    __builtin_amdgcn_s_wait_tensorcnt(0);
  }
  __syncthreads();

  float a_i = ai[(size_t)bi * Hd + hh];
  float we  = We[hh];
  const float* ajb = aj + (size_t)b * Nnode * Hd + hh;
  float acc = 0.0f;
#pragma unroll 4
  for (int j = 0; j < Nnode; ++j) {
    float x = a_i + ajb[(size_t)j * Hd] + jrow[j] * we;
    acc += silu_f(x);
  }
  S[(size_t)bi * Hd + hh] = acc;
}

// h = h * (1 + gamma[b,:]) + shift[b,:] ;  [gamma|shift] = gemb[b]@fw + fb
__global__ void film_kernel(float* __restrict__ h,
                            const float* __restrict__ gemb,
                            const float* __restrict__ fw,
                            const float* __restrict__ fb)
{
  int idx = blockIdx.x * blockDim.x + threadIdx.x;
  int hh = idx & (Hd - 1);
  int b  = idx >> 15;
  float g0 = gemb[b * 2], g1 = gemb[b * 2 + 1];
  float gamma = g0 * fw[hh]      + g1 * fw[2 * Hd + hh]      + fb[hh];
  float shift = g0 * fw[Hd + hh] + g1 * fw[2 * Hd + Hd + hh] + fb[Hd + hh];
  h[idx] = h[idx] * (1.0f + gamma) + shift;
}

// out[row] = softplus( sum_h silu(r1[row,h]) * w2[h] + b2 )
__global__ void readout_kernel(const float* __restrict__ r1,
                               const float* __restrict__ w2,
                               const float* __restrict__ b2,
                               float* __restrict__ out)
{
  __shared__ float red[Hd];
  int row = blockIdx.x;
  int hh  = threadIdx.x;
  red[hh] = silu_f(r1[(size_t)row * Hd + hh]) * w2[hh];
  __syncthreads();
  for (int s = Hd / 2; s > 0; s >>= 1) {
    if (hh < s) red[hh] += red[hh + s];
    __syncthreads();
  }
  if (hh == 0) {
    float z = red[0] + b2[0];
    out[row] = (z > 20.0f) ? z : log1pf(__expf(z));
  }
}

extern "C" void kernel_launch(void* const* d_in, const int* in_sizes, int n_in,
                              void* d_out, int out_size, void* d_ws, size_t ws_size,
                              hipStream_t stream)
{
  (void)in_sizes; (void)n_in; (void)out_size; (void)ws_size;

  const float* x_t     = (const float*)d_in[0];
  const float* t       = (const float*)d_in[1];
  const float* beta    = (const float*)d_in[2];
  const float* J_mat   = (const float*)d_in[3];
  const float* h_field = (const float*)d_in[4];
  const float* npw     = (const float*)d_in[5];
  const float* npb     = (const float*)d_in[6];
  const float* ge_w1   = (const float*)d_in[7];
  const float* ge_b1   = (const float*)d_in[8];
  const float* ge_w2   = (const float*)d_in[9];
  const float* ge_b2   = (const float*)d_in[10];
  const float* msg_w1  = (const float*)d_in[11];
  const float* msg_b1  = (const float*)d_in[12];
  const float* msg_w2  = (const float*)d_in[13];
  const float* msg_b2  = (const float*)d_in[14];
  const float* upd_w1  = (const float*)d_in[15];
  const float* upd_b1  = (const float*)d_in[16];
  const float* upd_w2  = (const float*)d_in[17];
  const float* upd_b2  = (const float*)d_in[18];
  const float* film_w  = (const float*)d_in[19];
  const float* film_b  = (const float*)d_in[20];
  const float* ro_w1   = (const float*)d_in[21];
  const float* ro_b1   = (const float*)d_in[22];
  const float* ro_w2   = (const float*)d_in[23];
  const float* ro_b2   = (const float*)d_in[24];

  const int ACT = Bn * Nnode * Hd;       // 131072
  float* h    = (float*)d_ws;
  float* buf1 = h + ACT;                  // ai / agg / r1
  float* buf2 = buf1 + ACT;               // aj / u
  float* S    = buf2 + ACT;               // silu sums
  float* gemb = S + ACT;                  // 8 floats

  const int M = Bn * Nnode;               // 1024 rows
  const int gemmBlocks = (M / 16) * (Hd / 16);   // 512 waves
  dim3 w32(32);

  init_h_kernel<<<ACT / 256, 256, 0, stream>>>(x_t, h_field, npw, npb, h);
  gemb_kernel<<<1, Hd, 0, stream>>>(t, beta, ge_w1, ge_b1, ge_w2, ge_b2, gemb);

  for (int l = 0; l < Ld; ++l) {
    const float* Wi  = msg_w1 + (size_t)l * 257 * Hd;
    const float* Wj  = Wi + 128 * Hd;
    const float* We  = Wi + 256 * Hd;
    const float* mb1 = msg_b1 + (size_t)l * Hd;
    const float* W2  = msg_w2 + (size_t)l * Hd * Hd;
    const float* mb2 = msg_b2 + (size_t)l * Hd;
    const float* Ua  = upd_w1 + (size_t)l * 2 * Hd * Hd;
    const float* Ub  = Ua + Hd * Hd;
    const float* ub1 = upd_b1 + (size_t)l * Hd;
    const float* Uw2 = upd_w2 + (size_t)l * Hd * Hd;
    const float* ub2 = upd_b2 + (size_t)l * Hd;
    const float* fw  = film_w + (size_t)l * 2 * 2 * Hd;
    const float* fb  = film_b + (size_t)l * 2 * Hd;

    // ai = h@Wi + msg_b1 ; aj = h@Wj
    gemm_wmma_f16<<<gemmBlocks, w32, 0, stream>>>(h, Wi, nullptr, nullptr,
        mb1, 1.0f, nullptr, buf1, 0);
    gemm_wmma_f16<<<gemmBlocks, w32, 0, stream>>>(h, Wj, nullptr, nullptr,
        nullptr, 0.0f, nullptr, buf2, 0);
    // S = sum_j silu(pre)   (only O(N^2) work left after sum-before-GEMM fusion)
    msg_sum_kernel<<<M, Hd, 0, stream>>>(buf1, buf2, J_mat, We, S);
    // agg = S@W2 + N*msg_b2
    gemm_wmma_f16<<<gemmBlocks, w32, 0, stream>>>(S, W2, nullptr, nullptr,
        mb2, (float)Nnode, nullptr, buf1, 0);
    // u = h@Ua + agg@Ub + upd_b1   (dual-GEMM accumulate)
    gemm_wmma_f16<<<gemmBlocks, w32, 0, stream>>>(h, Ua, buf1, Ub,
        ub1, 1.0f, nullptr, buf2, 0);
    // h = h + silu(u)@upd_w2 + upd_b2
    gemm_wmma_f16<<<gemmBlocks, w32, 0, stream>>>(buf2, Uw2, nullptr, nullptr,
        ub2, 1.0f, h, h, 1);
    // FiLM
    film_kernel<<<ACT / 256, 256, 0, stream>>>(h, gemb, fw, fb);
  }

  gemm_wmma_f16<<<gemmBlocks, w32, 0, stream>>>(h, ro_w1, nullptr, nullptr,
      ro_b1, 1.0f, nullptr, buf1, 0);
  readout_kernel<<<M, Hd, 0, stream>>>(buf1, ro_w2, ro_b2, (float*)d_out);
}
